// WaveNetSim_9990093931263
// MI455X (gfx1250) — matrix operand order repaired and checked by reading, not compiled, not run on hardware
//
#include <hip/hip_runtime.h>
#include <hip/hip_bf16.h>

// ---------------------------------------------------------------------------
// WaveNet 40-layer dilated conv stack on MI455X (gfx1250, wave32).
//
// Strategy: compose each 5-layer group (dilations s*{1,2,4,8,16}, s in {1,32})
// into a 32-tap filter; apply the 8 group filters as banded-Toeplitz GEMMs
// with v_wmma_f32_16x16x32_bf16, fully fused per time-chunk in LDS.
// Global traffic = 1x read + 1x write (~67 MB total) -> HBM bound ~3us;
// WMMA work ~8.6 GFLOP through the bf16 matrix pipe.
// Input staging uses global_load_async_to_lds_b128 (ASYNCcnt) when available.
// ---------------------------------------------------------------------------

typedef __attribute__((ext_vector_type(16))) __bf16       v16bf;
typedef __attribute__((ext_vector_type(8)))  float        v8f;
typedef __attribute__((ext_vector_type(8)))  unsigned int v8u;
typedef __attribute__((ext_vector_type(4)))  int          v4i;

#define BATCH    64
#define T_LEN    131072
#define CHUNK    4096
#define HALO     4092              // receptive field - 1
#define MAXLEN   (CHUNK + HALO)    // 8188 samples per chunk incl. halo
#define BUFSZ    10240             // MAXLEN + slack for over-reads by edge tiles
#define STG      4096              // f32 staging segment (16 KB LDS)
#define NTHREADS 256
#define NWAVES   8

// ---------------------------------------------------------------------------
// Kernel 1: compose the 8 group filters (32 taps each) + group biases.
// Group g covers layers g*5 .. g*5+4; relative dilations 1,2,4,8,16 in the
// (possibly 32-decimated) domain. f_new[k] = w0*f[k] + w1*f[k-rel].
// Bias recurrence: beta = (w0+w1)*beta + b.
// ws layout (floats): [0..255] coefs (8 x 32), [256..263] betas.
// ---------------------------------------------------------------------------
__global__ void build_filters(const float* __restrict__ W,
                              const float* __restrict__ b,
                              float* __restrict__ ws) {
  int g = threadIdx.x;
  if (g >= 8) return;
  float f[32];
#pragma unroll
  for (int k = 0; k < 32; ++k) f[k] = 0.0f;
  f[0] = 1.0f;
  float beta = 0.0f;
  for (int l = 0; l < 5; ++l) {
    int layer = g * 5 + l;
    float w0 = W[2 * layer];
    float w1 = W[2 * layer + 1];
    int rel = 1 << l;
    float nf[32];
#pragma unroll
    for (int k = 0; k < 32; ++k) {
      float lo = f[k];                                   // zero beyond old len
      float hi = (k >= rel) ? f[k - rel] : 0.0f;
      nf[k] = w0 * lo + w1 * hi;
    }
#pragma unroll
    for (int k = 0; k < 32; ++k) f[k] = nf[k];
    beta = (w0 + w1) * beta + b[layer];
  }
  for (int k = 0; k < 32; ++k) ws[g * 32 + k] = f[k];
  ws[256 + g] = beta;
}

// ---------------------------------------------------------------------------
// Pass with stride 1: y[t] = sum_j c[j] * h[t+j], j in [0,32).
// GEMM mapping: Y[m,n] = y[t0 + 16m + n]; A[m,k] = h[t0+16m+k] (data, K=64),
// B[k,n] = banded filter (c[k-n] on the band). Two 16x16x32 WMMAs chain K.
// ---------------------------------------------------------------------------
__device__ __forceinline__ void pass_s1(const __bf16* __restrict__ src,
                                        __bf16* __restrict__ dst,
                                        const float* __restrict__ c,
                                        float beta, int newLen) {
  const unsigned L = threadIdx.x & 31u;
  const int wave = threadIdx.x >> 5;
  const int half = (int)(L >> 4);   // K-subset selector for A/B 16-bit layouts
  const int lid  = (int)(L & 15u);

  // Build banded filter in B-layout (lanes 0-15: K=2v..; lanes 16-31: K=16+2v..)
  v16bf b1, b2;
#pragma unroll
  for (int v = 0; v < 8; ++v) {
#pragma unroll
    for (int e = 0; e < 2; ++e) {
      int k = half * 16 + 2 * v + e;   // K within [0,32)
      int n = lid;
      float f1 = (k >= n) ? c[k - n] : 0.0f;           // K-half 0: k' = k
      float f2 = (k <  n) ? c[32 + k - n] : 0.0f;      // K-half 1: k' = 32+k
      b1[2 * v + e] = (__bf16)f1;
      b2[2 * v + e] = (__bf16)f2;
    }
  }

  const unsigned int* s32p = (const unsigned int*)src;  // packed bf16 pairs
  const int nt = (newLen + 255) >> 8;
  for (int tile = wave; tile < nt; tile += NWAVES) {    // wave-uniform loop
    const int t0 = tile << 8;
    // A (data) per 16-bit A-matrix layout: row m = lid, packed K pairs.
    v8u au1, au2;
#pragma unroll
    for (int v = 0; v < 8; ++v) {
      int k0 = half * 16 + 2 * v;
      int idx = (t0 + 16 * lid + k0) >> 1;              // b32 = 2 bf16 taps
      au1[v] = s32p[idx];
      au2[v] = s32p[idx + 16];                          // K-half 1: +32 elems
    }
    v16bf a1 = __builtin_bit_cast(v16bf, au1);
    v16bf a2 = __builtin_bit_cast(v16bf, au2);
    v8f acc = {};
    acc = __builtin_amdgcn_wmma_f32_16x16x32_bf16(false, a1, false, b1,
                                                  (short)0, acc, false, false);
    acc = __builtin_amdgcn_wmma_f32_16x16x32_bf16(false, a2, false, b2,
                                                  (short)0, acc, false, false);
    const int tb = t0 + 16 * (half * 8) + lid;          // t for r = 0
    if (t0 + 256 <= newLen) {
      // Interior tile (wave-uniform): unguarded stores.
#pragma unroll
      for (int r = 0; r < 8; ++r)
        dst[tb + 16 * r] = (__bf16)(acc[r] + beta);
    } else {
#pragma unroll
      for (int r = 0; r < 8; ++r) {
        int t = tb + 16 * r;
        if (t < newLen) dst[t] = (__bf16)(acc[r] + beta);
      }
    }
  }
}

// ---------------------------------------------------------------------------
// Pass with stride 32: y[t] = sum_j c[j] * h[t+32j].
// GEMM mapping: Y[m,n] = y[t0 + 32m + n] (n = residue slice of 16);
// A[m,k] = banded filter (c[k-m] on the band), B[k,n] = h[t0+32k+n] (data).
// ---------------------------------------------------------------------------
__device__ __forceinline__ void pass_s32(const __bf16* __restrict__ src,
                                         __bf16* __restrict__ dst,
                                         const float* __restrict__ c,
                                         float beta, int newLen) {
  const unsigned L = threadIdx.x & 31u;
  const int wave = threadIdx.x >> 5;
  const int half = (int)(L >> 4);
  const int lid  = (int)(L & 15u);

  // Build banded filter in A-layout (row m = lid; K pairs, K = 16*half + 2v..).
  v16bf a1, a2;
#pragma unroll
  for (int v = 0; v < 8; ++v) {
    int k0 = half * 16 + 2 * v;
#pragma unroll
    for (int e = 0; e < 2; ++e) {
      int k = k0 + e;
      int m = lid;
      float f1 = (k >= m) ? c[k - m] : 0.0f;
      float f2 = (k <  m) ? c[32 + k - m] : 0.0f;
      a1[2 * v + e] = (__bf16)f1;
      a2[2 * v + e] = (__bf16)f2;
    }
  }

  const unsigned short* su = (const unsigned short*)src;
  const int ntu = (newLen + 511) >> 9;
  const int nt = ntu * 2;                                // 2 residue slices
  for (int tile = wave; tile < nt; tile += NWAVES) {
    const int t0 = ((tile >> 1) << 9) + ((tile & 1) << 4);
    // B (data), this code's contiguous K order: lanes 0-15 K=2v.., lanes 16-31 K=16+2v..
    v8u bu1, bu2;
#pragma unroll
    for (int v = 0; v < 8; ++v) {
      int klo = half * 16 + 2 * v;
      int base = t0 + lid + 32 * klo;                    // h[t0 + 32K + n]
      bu1[v] = (unsigned)su[base]        | ((unsigned)su[base + 32]   << 16);
      bu2[v] = (unsigned)su[base + 1024] | ((unsigned)su[base + 1056] << 16);
    }
    v16bf b1 = __builtin_bit_cast(v16bf, bu1);
    v16bf b2 = __builtin_bit_cast(v16bf, bu2);
    v8f acc = {};
    acc = __builtin_amdgcn_wmma_f32_16x16x32_bf16(false, a1, false, b1,
                                                  (short)0, acc, false, false);
    acc = __builtin_amdgcn_wmma_f32_16x16x32_bf16(false, a2, false, b2,
                                                  (short)0, acc, false, false);
    const int tb = t0 + 32 * (half * 8) + lid;           // t for r = 0
    if (t0 + 496 <= newLen) {
      // Interior tile (wave-uniform): unguarded stores.
#pragma unroll
      for (int r = 0; r < 8; ++r)
        dst[tb + 32 * r] = (__bf16)(acc[r] + beta);
    } else {
#pragma unroll
      for (int r = 0; r < 8; ++r) {
        int t = tb + 32 * r;
        if (t < newLen) dst[t] = (__bf16)(acc[r] + beta);
      }
    }
  }
}

// ---------------------------------------------------------------------------
// Kernel 2: fully fused 8-pass chunk processor. One workgroup = one chunk of
// 4096 outputs (+4092 halo), double-buffered bf16 in LDS plus a 16 KB f32
// async-staging segment (~58 KB static LDS total, well under the 320 KB WGP).
// ---------------------------------------------------------------------------
__global__ __launch_bounds__(NTHREADS)
void wavenet_fused(const float* __restrict__ x,
                   const float* __restrict__ ws,
                   float* __restrict__ out) {
  __shared__ __align__(16) __bf16 bufA[BUFSZ];
  __shared__ __align__(16) __bf16 bufB[BUFSZ];
  __shared__ __align__(16) float  stg[STG];
  __shared__ float cs[8 * 32];
  __shared__ float betas[8];

  const int tid = threadIdx.x;
  const int bb  = blockIdx.y;
  const int c0  = blockIdx.x * CHUNK;
  const float* xrow = x + (size_t)bb * T_LEN;

  cs[tid & 255] = ws[tid & 255];
  if (tid < 8) betas[tid] = ws[256 + tid];

#if defined(__HIP_DEVICE_COMPILE__) && \
    __has_builtin(__builtin_amdgcn_global_load_async_to_lds_b128)
  // Async DMA the f32 chunk (+halo) into LDS staging, 16 B per lane per op,
  // then convert to bf16. Two segments keep staging at 16 KB.
  for (int seg = 0; seg < MAXLEN; seg += STG) {
    const int segLen = (MAXLEN - seg < STG) ? (MAXLEN - seg) : STG;
    const int tg0 = c0 - HALO + seg;            // global t of stg[0]
    const int skip = (tg0 < 0) ? -tg0 : 0;      // causal zero pad (mult. of 4)
    for (int i = tid; i < skip; i += NTHREADS) stg[i] = 0.0f;
    const int nl = (segLen - skip) >> 2;        // 4 floats per b128
    for (int j = tid; j < nl; j += NTHREADS) {
      const float* gp = xrow + tg0 + skip + 4 * j;
      __builtin_amdgcn_global_load_async_to_lds_b128(
          (__attribute__((address_space(1))) v4i*)(unsigned long long)gp,
          (__attribute__((address_space(3))) v4i*)(unsigned)(unsigned long long)&stg[skip + 4 * j],
          0, 0);
    }
    asm volatile("s_wait_asynccnt 0" ::: "memory");
    __syncthreads();
    for (int i = tid; i < segLen; i += NTHREADS)
      bufA[seg + i] = (__bf16)stg[i];
    __syncthreads();                            // stg reused next segment
  }
  for (int i = MAXLEN + tid; i < BUFSZ; i += NTHREADS) bufA[i] = (__bf16)0.0f;
  for (int i = tid; i < BUFSZ; i += NTHREADS) bufB[i] = (__bf16)0.0f;
#else
  // Fallback: direct loads with on-the-fly convert.
  for (int i = tid; i < BUFSZ; i += NTHREADS) {
    float v = 0.0f;
    int t = c0 - HALO + i;
    if (i < MAXLEN && t >= 0) v = xrow[t];
    bufA[i] = (__bf16)v;
    bufB[i] = (__bf16)0.0f;
  }
#endif
  // Warm L2 for the next chunk of this row (global_prefetch_b8).
  if (c0 + CHUNK < T_LEN)
    __builtin_prefetch(&xrow[c0 + CHUNK + tid * 64], 0, 1);
  __syncthreads();

  const __bf16* src = bufA;
  __bf16*       dst = bufB;
  int len = MAXLEN;
  for (int g = 0; g < 8; ++g) {                 // 4 blocks x {stride1, stride32}
    const float* c = &cs[g * 32];
    float beta = betas[g];
    const int reach = (g & 1) ? 992 : 31;       // 31*s
    const int newLen = len - reach;
    if (g & 1) pass_s32(src, dst, c, beta, newLen);
    else       pass_s1 (src, dst, c, beta, newLen);
    __syncthreads();
    const __bf16* t = src; src = dst; dst = (__bf16*)t;
    len = newLen;
  }

  // src now holds y[c0 .. c0+CHUNK) at indices [0, CHUNK).
  for (int i = tid; i < CHUNK; i += NTHREADS)
    out[(size_t)bb * T_LEN + c0 + i] = (float)src[i];
}

// ---------------------------------------------------------------------------
extern "C" void kernel_launch(void* const* d_in, const int* in_sizes, int n_in,
                              void* d_out, int out_size, void* d_ws, size_t ws_size,
                              hipStream_t stream) {
  const float* x = (const float*)d_in[0];   // [64, 1, 131072]
  const float* W = (const float*)d_in[1];   // [40, 2]
  const float* b = (const float*)d_in[2];   // [40]
  float* ws = (float*)d_ws;                 // 264 floats used

  build_filters<<<dim3(1), dim3(32), 0, stream>>>(W, b, ws);

  dim3 grid(T_LEN / CHUNK, BATCH);          // 32 x 64 workgroups
  wavenet_fused<<<grid, dim3(NTHREADS), 0, stream>>>(x, ws, (float*)d_out);
}
